// HubCreator_59923383714407
// MI455X (gfx1250) — matrix operand order repaired and checked by reading, not compiled, not run on hardware
//
#include <hip/hip_runtime.h>
#include <hip/hip_bf16.h>
#include <math.h>

// Problem constants (match reference)
#define NSPOKE 524288
#define DIM    128      // D
#define DHID   256      // 2D
#define NB     128      // graphs
#define NH     64       // hubs/graph
#define KNN_K  4
#define NHUBS  (NB * NH)        // 8192

#define HP 272   // h-scratch row pitch in bf16 elems (256+16): 136 dwords == 8 mod 64 banks

typedef __attribute__((ext_vector_type(16))) __bf16 v16bf;
typedef __attribute__((ext_vector_type(8)))  float  v8f;

// ---------------------------------------------------------------------------
// Kernel 1: fused FFN (GEMM -> GELU -> GEMM) via bf16 WMMA + scatter-add.
// One wave processes a 16-row tile. All LDS fragment traffic is 32B/lane
// contiguous (2x ds_load_b128 per WMMA operand).
// ---------------------------------------------------------------------------
__global__ void ffn_scatter_kernel(const float* __restrict__ x,
                                   const int*   __restrict__ hub0,
                                   const int*   __restrict__ bidx,
                                   const float* __restrict__ W1,
                                   const float* __restrict__ b1,
                                   const float* __restrict__ W2,
                                   const float* __restrict__ b2,
                                   float* __restrict__ sums,
                                   int n_tiles)
{
    extern __shared__ char smem[];
    // W1 B-fragments: [nt=16][kb=4][lane=32][e=16] bf16  -> 64 KB
    __bf16* w1f = (__bf16*)smem;
    // W2 B-fragments: [nt=8][kb=8][lane=32][e=16] bf16   -> 64 KB
    __bf16* w2f = w1f + 16 * 4 * 32 * 16;
    // h scratch: [8 waves][16 rows][HP] bf16              -> 68 KB
    __bf16* hsb = w2f + 8 * 8 * 32 * 16;

    // ---- stage W1 into B-fragment order (fp32 -> bf16), once per block ----
    for (int f = threadIdx.x; f < 16 * 4 * 32 * 16; f += blockDim.x) {
        const int e  = f & 15;
        const int ln = (f >> 4) & 31;
        const int kb = (f >> 9) & 3;
        const int nt = f >> 11;
        const int K  = kb * 32 + (ln >> 4) * 8 + (e & 7) + (e >> 3) * 16;   // 0..127
        const int n  = nt * 16 + (ln & 15);            // 0..255
        w1f[f] = (__bf16)W1[K * DHID + n];
    }
    // ---- stage W2 into B-fragment order ----
    for (int f = threadIdx.x; f < 8 * 8 * 32 * 16; f += blockDim.x) {
        const int e  = f & 15;
        const int ln = (f >> 4) & 31;
        const int kb = (f >> 9) & 7;
        const int nt = f >> 12;
        const int K  = kb * 32 + (ln >> 4) * 8 + (e & 7) + (e >> 3) * 16;   // 0..255
        const int n  = nt * 16 + (ln & 15);            // 0..127
        w2f[f] = (__bf16)W2[K * DIM + n];
    }
    __syncthreads();

    const int wv   = threadIdx.x >> 5;
    const int lane = threadIdx.x & 31;
    const int half = lane >> 4;
    const int mrow = lane & 15;      // A row within tile
    const int col  = mrow;           // B/C column within n-tile
    __bf16* hs = hsb + wv * 16 * HP; // wave-private activation scratch

    const v16bf* w1v = (const v16bf*)w1f;
    const v16bf* w2v = (const v16bf*)w2f;

    // permuted column position within a 32-block (swap middle 8-groups):
    // makes GEMM2 A-fragments contiguous in LDS.
    const int poscol = col + (col < 8 ? 0 : 8);

    // tile-invariant biases (per-lane column slices)
    float bias1[16], bias2[8];
    #pragma unroll
    for (int nt = 0; nt < 16; ++nt) bias1[nt] = b1[nt * 16 + col];
    #pragma unroll
    for (int nt = 0; nt < 8; ++nt)  bias2[nt] = b2[nt * 16 + col];

    const int wave_global = blockIdx.x * (blockDim.x >> 5) + wv;
    const int wave_stride = gridDim.x * (blockDim.x >> 5);

    for (int tile = wave_global; tile < n_tiles; tile += wave_stride) {
        const int row0 = tile << 4;

        // ---- A fragments for GEMM1: wide global loads + cvt to bf16 ----
        v16bf a1[4];
        const float4* xr4 = (const float4*)(x + (size_t)(row0 + mrow) * DIM);
        #pragma unroll
        for (int kb = 0; kb < 4; ++kb) {
            const int base = kb * 8 + half * 2;     // float4 index of K base
            float4 p0 = xr4[base],     p1 = xr4[base + 1];
            float4 q0 = xr4[base + 4], q1 = xr4[base + 5];
            a1[kb][0]  = (__bf16)p0.x; a1[kb][1]  = (__bf16)p0.y;
            a1[kb][2]  = (__bf16)p0.z; a1[kb][3]  = (__bf16)p0.w;
            a1[kb][4]  = (__bf16)p1.x; a1[kb][5]  = (__bf16)p1.y;
            a1[kb][6]  = (__bf16)p1.z; a1[kb][7]  = (__bf16)p1.w;
            a1[kb][8]  = (__bf16)q0.x; a1[kb][9]  = (__bf16)q0.y;
            a1[kb][10] = (__bf16)q0.z; a1[kb][11] = (__bf16)q0.w;
            a1[kb][12] = (__bf16)q1.x; a1[kb][13] = (__bf16)q1.y;
            a1[kb][14] = (__bf16)q1.z; a1[kb][15] = (__bf16)q1.w;
        }

        // prefetch next tile's x rows behind the WMMAs
        const int ntile = tile + wave_stride;
        if (ntile < n_tiles)
            __builtin_prefetch(x + ((size_t)(ntile << 4) + mrow) * DIM, 0, 3);

        // ---- GEMM1: h1[16x256] = A * W1; +b1; exact GELU; store permuted ----
        #pragma unroll
        for (int nt = 0; nt < 16; ++nt) {
            v8f c = {};
            #pragma unroll
            for (int kb = 0; kb < 4; ++kb) {
                v16bf b = w1v[(nt * 4 + kb) * 32 + lane];   // one 32B LDS read
                c = __builtin_amdgcn_wmma_f32_16x16x32_bf16(
                        false, a1[kb], false, b, (short)0, c, false, false);
            }
            const float bias = bias1[nt];
            const int pos = (nt >> 1) * 32 + poscol + (nt & 1) * 8;
            #pragma unroll
            for (int i = 0; i < 8; ++i) {
                float v = c[i] + bias;
                v = 0.5f * v * (1.0f + erff(v * 0.70710678118654752f));
                const int m = half * 8 + i;
                hs[m * HP + pos] = (__bf16)v;
            }
        }

        // ---- A fragments for GEMM2: contiguous 32B LDS reads ----
        v16bf a2[8];
        #pragma unroll
        for (int kb = 0; kb < 8; ++kb)
            a2[kb] = *(const v16bf*)(hs + mrow * HP + kb * 32 + half * 16);

        // hub index for this lane's 8 C rows (wide int4 loads)
        const int rbase = row0 + half * 8;
        int4 bi0 = *(const int4*)(bidx + rbase);
        int4 bi1 = *(const int4*)(bidx + rbase + 4);
        int4 hi0 = *(const int4*)(hub0 + rbase);
        int4 hi1 = *(const int4*)(hub0 + rbase + 4);
        int g[8] = { bi0.x * NH + hi0.x, bi0.y * NH + hi0.y,
                     bi0.z * NH + hi0.z, bi0.w * NH + hi0.w,
                     bi1.x * NH + hi1.x, bi1.y * NH + hi1.y,
                     bi1.z * NH + hi1.z, bi1.w * NH + hi1.w };

        // ---- GEMM2: out[16x128] = h1 * W2; +b2; scatter-add into hubs ----
        #pragma unroll
        for (int nt = 0; nt < 8; ++nt) {
            v8f c = {};
            #pragma unroll
            for (int kb = 0; kb < 8; ++kb) {
                v16bf b = w2v[(nt * 8 + kb) * 32 + lane];   // one 32B LDS read
                c = __builtin_amdgcn_wmma_f32_16x16x32_bf16(
                        false, a2[kb], false, b, (short)0, c, false, false);
            }
            const float bias = bias2[nt];
            const int n0 = nt * 16;
            #pragma unroll
            for (int i = 0; i < 8; ++i)
                atomicAdd(&sums[(size_t)g[i] * DIM + n0 + col], c[i] + bias);
        }
    }
}

// ---------------------------------------------------------------------------
// Kernel 2: per-hub counts
// ---------------------------------------------------------------------------
__global__ void count_kernel(const int* __restrict__ hub0,
                             const int* __restrict__ bidx,
                             float* __restrict__ cnt, int n)
{
    int i = blockIdx.x * blockDim.x + threadIdx.x;
    if (i < n) atomicAdd(&cnt[bidx[i] * NH + hub0[i]], 1.0f);
}

// ---------------------------------------------------------------------------
// Kernel 3: hub_features = sums / max(cnt, 1)
// ---------------------------------------------------------------------------
__global__ void finalize_kernel(const float* __restrict__ sums,
                                const float* __restrict__ cnt,
                                float* __restrict__ out, int total)
{
    int i = blockIdx.x * blockDim.x + threadIdx.x;
    if (i < total) out[i] = sums[i] / fmaxf(cnt[i >> 7], 1.0f);
}

// ---------------------------------------------------------------------------
// Kernel 4: per-graph 64x64 pairwise d^2 + top-4 nearest (stable, asc scan)
// ---------------------------------------------------------------------------
__global__ void knn_kernel(const float* __restrict__ hubf,
                           int* __restrict__ knn)
{
    __shared__ float hf[NH * DIM];   // 32 KB
    const int b = blockIdx.x;
    for (int idx = threadIdx.x; idx < NH * DIM; idx += blockDim.x)
        hf[idx] = hubf[(size_t)b * NH * DIM + idx];
    __syncthreads();

    const int h = threadIdx.x;
    if (h < NH) {
        int chosen0 = -1, chosen1 = -1, chosen2 = -1, chosen3 = -1;
        #pragma unroll
        for (int k = 0; k < KNN_K; ++k) {
            float best = 3.4e38f;
            int bestj = 0;
            for (int j = 0; j < NH; ++j) {
                if (j == chosen0 || j == chosen1 || j == chosen2 || j == chosen3)
                    continue;
                float d = 0.0f;
                for (int dd = 0; dd < DIM; ++dd) {
                    float t = hf[h * DIM + dd] - hf[j * DIM + dd];
                    d = fmaf(t, t, d);
                }
                if (d < best) { best = d; bestj = j; }  // asc scan => stable
            }
            if (k == 0) chosen0 = bestj;
            else if (k == 1) chosen1 = bestj;
            else if (k == 2) chosen2 = bestj;
            else chosen3 = bestj;
            knn[((size_t)b * NH + h) * KNN_K + k] = bestj;
        }
    }
}

// ---------------------------------------------------------------------------
// Kernel 5: emit edge index [2, N*K] (values are small ints, exact in f32)
// ---------------------------------------------------------------------------
__global__ void edge_kernel(const int* __restrict__ hub0,
                            const int* __restrict__ bidx,
                            const int* __restrict__ knn,
                            float* __restrict__ eout, int n)
{
    int i = blockIdx.x * blockDim.x + threadIdx.x;
    if (i < n) {
        const int b = bidx[i];
        const int* kr = knn + ((size_t)b * NH + hub0[i]) * KNN_K;
        const size_t NK = (size_t)n * KNN_K;
        #pragma unroll
        for (int k = 0; k < KNN_K; ++k) {
            eout[(size_t)i * KNN_K + k]      = (float)i;
            eout[NK + (size_t)i * KNN_K + k] = (float)(kr[k] + b * NH);
        }
    }
}

// ---------------------------------------------------------------------------
extern "C" void kernel_launch(void* const* d_in, const int* in_sizes, int n_in,
                              void* d_out, int out_size, void* d_ws, size_t ws_size,
                              hipStream_t stream)
{
    const float* x    = (const float*)d_in[0];
    const int*   hub0 = (const int*)  d_in[1];
    const int*   bidx = (const int*)  d_in[2];
    const float* W1   = (const float*)d_in[3];
    const float* b1   = (const float*)d_in[4];
    const float* W2   = (const float*)d_in[5];
    const float* b2   = (const float*)d_in[6];
    float* out = (float*)d_out;

    // workspace layout
    float* sums = (float*)d_ws;                        // NHUBS*DIM fp32
    float* cnt  = sums + (size_t)NHUBS * DIM;          // NHUBS fp32
    int*   knn  = (int*)(cnt + NHUBS);                 // NHUBS*K int

    hipMemsetAsync(d_ws, 0,
                   ((size_t)NHUBS * DIM + NHUBS) * sizeof(float), stream);

    const int n_tiles = NSPOKE / 16;                   // 32768
    const size_t shmem =
        (size_t)(16 * 4 * 32 * 16 + 8 * 8 * 32 * 16 + 8 * 16 * HP) * sizeof(__bf16);
    ffn_scatter_kernel<<<512, 256, shmem, stream>>>(
        x, hub0, bidx, W1, b1, W2, b2, sums, n_tiles);

    count_kernel<<<(NSPOKE + 255) / 256, 256, 0, stream>>>(hub0, bidx, cnt, NSPOKE);

    finalize_kernel<<<(NHUBS * DIM + 255) / 256, 256, 0, stream>>>(
        sums, cnt, out, NHUBS * DIM);

    knn_kernel<<<NB, 64, 0, stream>>>(out, knn);

    edge_kernel<<<(NSPOKE + 255) / 256, 256, 0, stream>>>(
        hub0, bidx, knn, out + (size_t)NHUBS * DIM, NSPOKE);
}
